// Network_1872605741517
// MI455X (gfx1250) — matrix operand order repaired and checked by reading, not compiled, not run on hardware
//
#include <hip/hip_runtime.h>
#include <hip/hip_bf16.h>
#include <math.h>

// ===========================================================================
// RandLA-Net-style point network for MI455X (gfx1250, wave32, WMMA + TDM).
// - All channel GEMMs: v_wmma_f32_16x16x32_f16, 16(cout)x16(points) per wave.
// - Weights pre-packed to f16 A-fragment layout (wprep) -> the GEMM hot loop
//   loads its A fragment with 2x global_load_b128 (no cvt/select/addr math).
// - k-loop split into full slices (pointer-bump addressing) + clamped tail.
// - Branchless float4 LDS staging, conflict-free [16][33] layout; per-wave
//   sync via "s_wait_dscnt 0" (same-wave LDS ops are in order on CDNA5).
// - Decoder skip copies via Tensor Data Mover (tensor_load_to_lds /
//   tensor_store_from_lds + s_wait_tensorcnt), D# per ISA 8.3/8.4.
// NOTE: every GEMM launch has M % 64 == 0, so no column guards anywhere.
// ===========================================================================

typedef __attribute__((ext_vector_type(16))) _Float16 v16h;
typedef __attribute__((ext_vector_type(8)))  float    v8f;
typedef unsigned int       u32;
typedef unsigned long long u64;
typedef u32 uv4 __attribute__((ext_vector_type(4)));
typedef int iv4 __attribute__((ext_vector_type(4)));
typedef int iv8 __attribute__((ext_vector_type(8)));

#define WPB 4        // waves per block (4 x wave32 = 128 threads)
#define NEGS 0.2f

__device__ __forceinline__ float lrelu(float x) { return x > 0.f ? x : NEGS * x; }
__device__ __forceinline__ void wave_lds_fence() {
    asm volatile("s_wait_dscnt 0" ::: "memory");
}

// ---------------------------------------------------------------------------
// Weight pre-pack: f32 W[Cout][Cin] -> f16 A-fragments, layout
//   Wf[((oT*ksteps + kk)*32 + lane)*16 + h]  k = 16*(lane/16) + h (contiguous, same as B).
// Zero-padding for cA >= Cin / orow >= Cout folded in here (once), so the
// GEMM hot loop has no selects.
// ---------------------------------------------------------------------------
__global__ void wprep_kernel(const float* __restrict__ W, _Float16* __restrict__ Wf,
                             int Cin, int Cout, int oTiles, int ksteps)
{
    const int tid   = blockIdx.x * blockDim.x + threadIdx.x;
    const int total = oTiles * ksteps * 32;
    if (tid >= total) return;
    const int lane = tid & 31;
    const int kk   = (tid >> 5) % ksteps;
    const int oT   = tid / (32 * ksteps);
    const int col  = lane & 15;
    const int hi   = lane >> 4;
    const int akb  = hi ? 8 : 0;
    const int orow  = oT * 16 + col;
    const int orowc = orow < (Cout - 1) ? orow : (Cout - 1);
    const int c0 = kk << 5;
    v16h frag = {};
#pragma unroll
    for (int h = 0; h < 16; ++h) {
        const int kA  = 2 * akb + h;
        const int cA  = c0 + kA;
        const int cAc = cA < (Cin - 1) ? cA : (Cin - 1);
        const float wv = W[(size_t)orowc * Cin + cAc];
        frag[h] = (cA < Cin && orow < Cout) ? (_Float16)wv : (_Float16)0.f;
    }
    *reinterpret_cast<v16h*>(Wf + (size_t)tid * 16) = frag;
}

// ---------------------------------------------------------------------------
// Generic fused conv_bn GEMM (pre-packed weights):
//   Y[b, c0o+o, m] = act( (sum_c W[o,c] * X[b, c0i+c, m]) * g[o] + beta[o] )
// ---------------------------------------------------------------------------
__global__ void __launch_bounds__(128)
gemm_bn_kernel(const _Float16* __restrict__ Wf,
               const float* __restrict__ gamma,
               const float* __restrict__ beta,
               const float* __restrict__ X,
               float* __restrict__ Y,
               int Cin, int Cout, int M,
               int CtI, int c0i, int CtO, int c0o, int act)
{
    __shared__ float lds[WPB][16][33];   // [wave][col][chan]; stride 33: conflict-free
    const int wave = threadIdx.x >> 5;
    const int lane = threadIdx.x & 31;
    const int b    = blockIdx.z;
    const int oT   = blockIdx.y;
    const int m0   = (blockIdx.x * WPB + wave) * 16;

    const int col = lane & 15;
    const int hi  = lane >> 4;
    const size_t xbase = ((size_t)b * CtI + c0i) * (size_t)M;
    const int ksteps = (Cin + 31) >> 5;
    const int kfull  = Cin >> 5;
    const int krem   = Cin & 31;
    const _Float16* wfrag = Wf + ((size_t)oT * ksteps * 32 + lane) * 16;

    v8f acc = {};
    const float* pX = X + xbase + (size_t)lane * M + m0;   // row c = lane
    for (int kk = 0; kk < kfull; ++kk) {
        // ---- stage 32(chan) x 16(col) f32 tile, pointer-bump addressing ----
        {
            const float4* p = reinterpret_cast<const float4*>(pX);
#pragma unroll
            for (int i4 = 0; i4 < 4; ++i4) {
                float4 v = p[i4];
                lds[wave][4 * i4 + 0][lane] = v.x;
                lds[wave][4 * i4 + 1][lane] = v.y;
                lds[wave][4 * i4 + 2][lane] = v.z;
                lds[wave][4 * i4 + 3][lane] = v.w;
            }
            if (kk + 1 < kfull)                       // prefetch next full slice
                __builtin_prefetch(pX + (size_t)32 * M, 0, 1);
            pX += (size_t)32 * M;
        }
        wave_lds_fence();
        const v16h aF = *reinterpret_cast<const v16h*>(wfrag + (size_t)kk * 32 * 16);
        v16h bF = {};
#pragma unroll
        for (int h = 0; h < 16; ++h)
            bF[h] = (_Float16)lds[wave][col][(hi ? 16 : 0) + h];
        acc = __builtin_amdgcn_wmma_f32_16x16x32_f16(
            false, aF, false, bF, (short)0, acc, false, false);
        wave_lds_fence();                             // WAR before restage
    }
    if (krem) {                                       // clamped tail slice
        const int c  = (kfull << 5) + lane;
        const int cc = c < (Cin - 1) ? c : (Cin - 1);
        const float4* p = reinterpret_cast<const float4*>(
            X + xbase + (size_t)cc * M + m0);
#pragma unroll
        for (int i4 = 0; i4 < 4; ++i4) {
            float4 v = p[i4];
            lds[wave][4 * i4 + 0][lane] = v.x;
            lds[wave][4 * i4 + 1][lane] = v.y;
            lds[wave][4 * i4 + 2][lane] = v.z;
            lds[wave][4 * i4 + 3][lane] = v.w;
        }
        wave_lds_fence();
        const v16h aF = *reinterpret_cast<const v16h*>(wfrag + (size_t)kfull * 32 * 16);
        v16h bF = {};
#pragma unroll
        for (int h = 0; h < 16; ++h)
            bF[h] = (_Float16)lds[wave][col][(hi ? 16 : 0) + h];
        acc = __builtin_amdgcn_wmma_f32_16x16x32_f16(
            false, aF, false, bF, (short)0, acc, false, false);
    }
    // ---- epilogue: BN + leaky-relu, channel-offset store ----
    const size_t ybase = ((size_t)b * CtO + c0o) * (size_t)M;
    const int m = m0 + col;
#pragma unroll
    for (int r = 0; r < 8; ++r) {
        const int oo = oT * 16 + r + (hi ? 8 : 0);
        if (oo < Cout) {
            float y = acc[r];
            if (gamma) y = y * gamma[oo] + beta[oo];
            if (act)   y = lrelu(y);
            Y[ybase + (size_t)oo * M + m] = y;
        }
    }
}

// ---------------------------------------------------------------------------
// Fused relf (dist, rel, tile, nxyz -> 10 chans) + conv_bn GEMM (10 -> Cout),
// pre-packed weights (ksteps == 1). One wave tile == one point n.
// ---------------------------------------------------------------------------
__global__ void __launch_bounds__(128)
gemm_relf_bn_kernel(const _Float16* __restrict__ Wf,
                    const float* __restrict__ gamma,
                    const float* __restrict__ beta,
                    const float* __restrict__ xyz,   // [B][N][3]
                    const int*   __restrict__ nidx,  // [B][N][16]
                    float* __restrict__ Y,
                    int Cout, int N, int CtO, int c0o, int act)
{
    __shared__ float lds[WPB][16][33];
    const int wave = threadIdx.x >> 5;
    const int lane = threadIdx.x & 31;
    const int b    = blockIdx.z;
    const int oT   = blockIdx.y;
    const int n    = blockIdx.x * WPB + wave;

    if (lane < 16) {
        const int    k  = lane;
        const int    j  = nidx[((size_t)b * N + n) * 16 + k];
        const float* xb = xyz + (size_t)b * N * 3;
        const float px = xb[(size_t)n * 3 + 0];
        const float py = xb[(size_t)n * 3 + 1];
        const float pz = xb[(size_t)n * 3 + 2];
        const float qx = xb[(size_t)j * 3 + 0];
        const float qy = xb[(size_t)j * 3 + 1];
        const float qz = xb[(size_t)j * 3 + 2];
        const float rx = px - qx, ry = py - qy, rz = pz - qz;
        const float d  = sqrtf(rx * rx + ry * ry + rz * rz);
        const float vals[10] = { d, rx, ry, rz, px, py, pz, qx, qy, qz };
#pragma unroll
        for (int c = 0; c < 10; ++c) lds[wave][k][c] = vals[c];
#pragma unroll
        for (int c = 10; c < 32; ++c) lds[wave][k][c] = 0.f;
    }
    wave_lds_fence();

    const int col = lane & 15;
    const int hi  = lane >> 4;
    const v16h aF = *reinterpret_cast<const v16h*>(Wf + ((size_t)oT * 32 + lane) * 16);
    v16h bF = {};
#pragma unroll
    for (int h = 0; h < 16; ++h)
        bF[h] = (_Float16)lds[wave][col][(hi ? 16 : 0) + h];
    v8f acc = {};
    acc = __builtin_amdgcn_wmma_f32_16x16x32_f16(
        false, aF, false, bF, (short)0, acc, false, false);

    const size_t M = (size_t)N * 16;
    const size_t ybase = ((size_t)b * CtO + c0o) * M;
    const int m = n * 16 + col;
#pragma unroll
    for (int r = 0; r < 8; ++r) {
        const int oo = oT * 16 + r + (hi ? 8 : 0);
        if (oo < Cout) {
            float y = acc[r];
            if (gamma) y = y * gamma[oo] + beta[oo];
            if (act)   y = lrelu(y);
            Y[ybase + (size_t)oo * M + m] = y;
        }
    }
}

// ---------------------------------------------------------------------------
// TDM slab copy: contiguous 64KB chunks, global -> LDS -> global via the
// Tensor Data Mover (D# per ISA 8.3/8.4: count=1, type=2, data_size=4B,
// 2D tile 4096 x 4 dwords, tensor_dim0_stride = 4096).
// ---------------------------------------------------------------------------
#define TDM_COLS 4096u      // dwords per row
#define TDM_ROWS 4u         // rows per chunk -> 16384 dwords = 64 KB

__device__ __forceinline__ uv4 tdm_group0(u64 ga, u32 lds_addr)
{
    uv4 g;
    g.x = 1u;                                           // count=1
    g.y = lds_addr;                                     // lds_addr (bytes)
    g.z = (u32)ga;                                      // global_addr[31:0]
    g.w = (u32)((ga >> 32) & 0x1FFFFFFu) | (2u << 30);  // addr[56:32] | type=2
    return g;
}
__device__ __forceinline__ iv8 tdm_group1_2d(u32 dim0, u32 dim1, u32 t0, u32 t1)
{
    iv8 g;
    g[0] = (int)(2u << 16);                              // data_size = 4 bytes
    g[1] = (int)((dim0 & 0xFFFFu) << 16);                // tensor_dim0[15:0]
    g[2] = (int)(((dim0 >> 16) & 0xFFFFu) | ((dim1 & 0xFFFFu) << 16));
    g[3] = (int)(((dim1 >> 16) & 0xFFFFu) | ((t0 & 0xFFFFu) << 16));  // tile_dim0
    g[4] = (int)(t1 & 0xFFFFu);                          // tile_dim1 (tile_dim2=0)
    g[5] = (int)dim0;                                    // tensor_dim0_stride lo
    g[6] = 0;
    g[7] = 0;
    return g;
}

__global__ void __launch_bounds__(32)
tdm_copy_kernel(const float* __restrict__ src, float* __restrict__ dst,
                int Cs, int N, int CtO, int c0o)
{
    __shared__ float smem[TDM_ROWS * TDM_COLS];
    const int    b     = blockIdx.y;
    const size_t chunk = (size_t)blockIdx.x * (TDM_ROWS * TDM_COLS);
    const float* s = src + ((size_t)b * Cs) * (size_t)N + chunk;
    float*       d = dst + ((size_t)b * CtO + c0o) * (size_t)N + chunk;
    const u32 lds_addr = (u32)(size_t)(void*)smem;   // flat LDS addr: low 32 = offset

#if __has_builtin(__builtin_amdgcn_tensor_load_to_lds)
    const iv4 z4 = { 0, 0, 0, 0 };
    const uv4 gl = tdm_group0((u64)(size_t)s, lds_addr);
    const uv4 gs = tdm_group0((u64)(size_t)d, lds_addr);
    const iv8 g1 = tdm_group1_2d(TDM_COLS, TDM_ROWS, TDM_COLS, TDM_ROWS);
#if defined(__clang_major__) && __clang_major__ >= 23
    const iv8 z8 = { 0, 0, 0, 0, 0, 0, 0, 0 };
    __builtin_amdgcn_tensor_load_to_lds(gl, g1, z4, z4, z8, 0);
    __builtin_amdgcn_s_wait_tensorcnt((short)0);
    __builtin_amdgcn_tensor_store_from_lds(gs, g1, z4, z4, z8, 0);
#else
    __builtin_amdgcn_tensor_load_to_lds(gl, g1, z4, z4, 0);
    __builtin_amdgcn_s_wait_tensorcnt((short)0);
    __builtin_amdgcn_tensor_store_from_lds(gs, g1, z4, z4, 0);
#endif
    __builtin_amdgcn_s_wait_tensorcnt((short)0);
#else
    for (u32 i = threadIdx.x; i < TDM_ROWS * TDM_COLS; i += 32) d[i] = s[i];
#endif
}

// ---------------------------------------------------------------------------
// Gather neighbors: out[b, c0o+c, n, k] = src[b, c, idx[b,n,k]]
// ---------------------------------------------------------------------------
__global__ void gather_nb_kernel(const float* __restrict__ src,
                                 const int* __restrict__ idx,
                                 float* __restrict__ out,
                                 int Cs, int Nout, int Ns, int Kk,
                                 int CtO, int c0o, int total)
{
    const int tid = blockIdx.x * blockDim.x + threadIdx.x;
    if (tid >= total) return;
    const int k = tid % Kk;
    const int n = (tid / Kk) % Nout;
    const int b = tid / (Kk * Nout);
    const int j = idx[tid];
    const float* s = src + ((size_t)b * Cs) * Ns + j;
    float* o = out + (((size_t)b * CtO + c0o) * Nout + n) * (size_t)Kk + k;
    const size_t os = (size_t)Nout * Kk;
    for (int c = 0; c < Cs; ++c)
        o[(size_t)c * os] = s[(size_t)c * Ns];
}

// ---------------------------------------------------------------------------
// Attention pool reduce: agg[b,c,n] = sum_k fs * softmax_k(att), K = 16.
// ---------------------------------------------------------------------------
__global__ void att_pool_reduce_kernel(const float* __restrict__ fs,
                                       const float* __restrict__ att,
                                       float* __restrict__ agg, int total)
{
    const int tid = blockIdx.x * blockDim.x + threadIdx.x;
    if (tid >= total) return;
    const float4* pa = reinterpret_cast<const float4*>(att + (size_t)tid * 16);
    const float4* pf = reinterpret_cast<const float4*>(fs  + (size_t)tid * 16);
    float a[16], f[16];
#pragma unroll
    for (int i = 0; i < 4; ++i) {
        float4 va = pa[i];
        a[4 * i + 0] = va.x; a[4 * i + 1] = va.y; a[4 * i + 2] = va.z; a[4 * i + 3] = va.w;
        float4 vf = pf[i];
        f[4 * i + 0] = vf.x; f[4 * i + 1] = vf.y; f[4 * i + 2] = vf.z; f[4 * i + 3] = vf.w;
    }
    float mx = a[0];
#pragma unroll
    for (int i = 1; i < 16; ++i) mx = fmaxf(mx, a[i]);
    float s = 0.f, dot = 0.f;
#pragma unroll
    for (int i = 0; i < 16; ++i) {
        const float e = __expf(a[i] - mx);
        s += e; dot += e * f[i];
    }
    agg[tid] = dot / s;
}

// ---------------------------------------------------------------------------
// random_sample: out[b,c,m] = max_k src[b,c, sub[b,m,k]]
// ---------------------------------------------------------------------------
__global__ void maxpool_kernel(const float* __restrict__ src,
                               const int* __restrict__ sub,
                               float* __restrict__ out,
                               int C, int Ns, int M, int Kk, int total)
{
    const int tid = blockIdx.x * blockDim.x + threadIdx.x;
    if (tid >= total) return;
    const int m = tid % M;
    const int c = (tid / M) % C;
    const int b = tid / (M * C);
    const int* ip = sub + ((size_t)b * M + m) * Kk;
    const float* s = src + ((size_t)b * C + c) * (size_t)Ns;
    float mx = -3.402823466e38f;
    for (int k = 0; k < Kk; ++k) mx = fmaxf(mx, s[ip[k]]);
    out[((size_t)b * C + c) * (size_t)M + m] = mx;
}

__global__ void add_leaky_kernel(const float* __restrict__ a,
                                 const float* __restrict__ b,
                                 float* __restrict__ out, size_t n)
{
    const size_t tid = (size_t)blockIdx.x * blockDim.x + threadIdx.x;
    if (tid >= n) return;
    out[tid] = lrelu(a[tid] + b[tid]);
}

// out[b, c0o+c, n] = src[b, c, n]  (fallback channel-slice copy)
__global__ void copy_slice_kernel(const float* __restrict__ src,
                                  float* __restrict__ out,
                                  int Cs, int N, int CtO, int c0o, int total)
{
    const int tid = blockIdx.x * blockDim.x + threadIdx.x;
    if (tid >= total) return;
    const int n = tid % N;
    const int c = (tid / N) % Cs;
    const int b = tid / (N * Cs);
    out[((size_t)b * CtO + c0o + c) * (size_t)N + n] =
        src[((size_t)b * Cs + c) * (size_t)N + n];
}

// ===========================================================================
// Host orchestration
// ===========================================================================
extern "C" void kernel_launch(void* const* d_in, const int* in_sizes, int n_in,
                              void* d_out, int out_size, void* d_ws, size_t ws_size,
                              hipStream_t stream)
{
    (void)in_sizes; (void)n_in; (void)out_size; (void)ws_size;
    static const int Bb = 4, Kn = 16;
    static const int DOUT[4] = { 16, 64, 128, 256 };
    static const int NSv[5]  = { 32768, 8192, 2048, 512, 128 };

    // ---- inputs (setup_inputs dict order) ----
    const float* features = (const float*)d_in[0];
    const float* xyz[4]; const int* neigh[4]; const int* sub[4]; const int* interp[4];
    for (int i = 0; i < 4; ++i) {
        xyz[i]    = (const float*)d_in[1 + 4 * i];
        neigh[i]  = (const int*)  d_in[2 + 4 * i];
        sub[i]    = (const int*)  d_in[3 + 4 * i];
        interp[i] = (const int*)  d_in[4 + 4 * i];
    }
    // ---- params: jax pytree flatten order (dict keys sorted) ----
    struct CB { const float *b, *g, *w; };
    auto cb = [&](int i) {
        return CB{ (const float*)d_in[i], (const float*)d_in[i + 1], (const float*)d_in[i + 2] };
    };
    CB dec_p[4];
    for (int j = 0; j < 4; ++j) dec_p[j] = cb(17 + 3 * j);
    CB dec0_p = cb(29);
    const float* att1_fc[4]; CB att1_mlp[4];
    const float* att2_fc[4]; CB att2_mlp[4];
    CB lfa1[4], lfa2[4], mlp1_p[4], mlp2_p[4], short_p[4];
    for (int i = 0; i < 4; ++i) {
        const int e = 32 + 23 * i;
        att1_fc[i]  = (const float*)d_in[e + 0];
        att1_mlp[i] = cb(e + 1);
        att2_fc[i]  = (const float*)d_in[e + 4];
        att2_mlp[i] = cb(e + 5);
        lfa1[i]     = cb(e + 8);
        lfa2[i]     = cb(e + 11);
        mlp1_p[i]   = cb(e + 14);
        mlp2_p[i]   = cb(e + 17);
        short_p[i]  = cb(e + 20);
    }
    CB fc0_p = cb(124);

    // ---- workspace carve-up (floats); total ~560 MB ----
    float* ws = (float*)d_ws;
    size_t off = 0;
    auto take = [&](size_t n) { float* p = ws + off; off += (n + 63) & ~(size_t)63; return p; };
    float* ENCp[5];
    ENCp[0] = take((size_t)Bb * 32  * 32768);  // fe0
    ENCp[1] = take((size_t)Bb * 32  * 8192);   // fs0
    ENCp[2] = take((size_t)Bb * 128 * 2048);   // fs1
    ENCp[3] = take((size_t)Bb * 256 * 512);    // fs2
    ENCp[4] = take((size_t)Bb * 512 * 128);    // fs3
    float* X0    = take((size_t)Bb * 8 * 32768);
    float* Fbuf  = take(1048576);
    float* CAT1  = take(33554432);
    float* CAT2  = take(33554432);
    float* SCORE = take(33554432);
    float* AGG   = take(2097152);
    float* FAGG  = take(1048576);
    float* BBv   = take(2097152);
    float* T1    = take(4194304);
    float* SCb   = take(4194304);
    float* FE    = take(4194304);
    float* CATIN = take(8388608);
    float* DF1   = take(1048576);
    float* DF2   = take(1048576);
    _Float16* WF = (_Float16*)take(262144);    // 524288 halves (max frag 262144)

    // prep + gemm launchers (sequential on stream -> WF reuse is safe)
    auto prep = [&](const float* W, int Cin, int Cout) {
        const int oT = (Cout + 15) / 16, ks = (Cin + 31) / 32;
        const int tot = oT * ks * 32;
        wprep_kernel<<<(tot + 255) / 256, 256, 0, stream>>>(W, WF, Cin, Cout, oT, ks);
    };
    auto gemm = [&](const float* W, const float* g, const float* be,
                    const float* X, float* Y, int Cin, int Cout, int M,
                    int CtI, int c0i, int CtO, int c0o, int act) {
        prep(W, Cin, Cout);
        dim3 grid((M + 63) / 64, (Cout + 15) / 16, Bb);
        gemm_bn_kernel<<<grid, 128, 0, stream>>>(WF, g, be, X, Y, Cin, Cout, M,
                                                 CtI, c0i, CtO, c0o, act);
    };
    auto copy_slab = [&](const float* src, float* dst, int Cs, int N, int CtO, int c0o) {
        const size_t perB = (size_t)Cs * N;
        if (perB % (TDM_ROWS * TDM_COLS) == 0) {
            dim3 grid((unsigned)(perB / (TDM_ROWS * TDM_COLS)), Bb);
            tdm_copy_kernel<<<grid, 32, 0, stream>>>(src, dst, Cs, N, CtO, c0o);
        } else {
            const int tot = Bb * Cs * N;
            copy_slice_kernel<<<(tot + 255) / 256, 256, 0, stream>>>(
                src, dst, Cs, N, CtO, c0o, tot);
        }
    };

    // ---- fc0: 6 -> 8, BN + leaky ----
    gemm(fc0_p.w, fc0_p.g, fc0_p.b, features, X0, 6, 8, 32768, 6, 0, 8, 0, 1);

    // ---- encoder ----
    for (int i = 0; i < 4; ++i) {
        const int N = NSv[i], d = DOUT[i], h = d / 2;
        const int din = (i == 0) ? 8 : 2 * DOUT[i - 1];
        const float* fin = (i == 0) ? X0 : ENCp[i];
        const int M = N * Kn;
        const int totNK = Bb * N * Kn;

        // f = mlp1(feat)  (din -> h) over N points
        gemm(mlp1_p[i].w, mlp1_p[i].g, mlp1_p[i].b, fin, Fbuf,
             din, h, N, din, 0, h, 0, 1);
        // f_xyz = lfa.mlp1(relf) fused -> CAT1 channels [h, d)
        {
            prep(lfa1[i].w, 10, h);
            dim3 grid(N / WPB, (h + 15) / 16, Bb);
            gemm_relf_bn_kernel<<<grid, 128, 0, stream>>>(
                WF, lfa1[i].g, lfa1[i].b, xyz[i], neigh[i],
                CAT1, h, N, d, h, 1);
        }
        // f_nb = gather(f) -> CAT1 channels [0, h)
        gather_nb_kernel<<<(totNK + 255) / 256, 256, 0, stream>>>(
            Fbuf, neigh[i], CAT1, h, N, N, Kn, d, 0, totNK);
        // scores1 = att1.fc(f_cat)   (d -> d, no BN/act)
        gemm(att1_fc[i], nullptr, nullptr, CAT1, SCORE, d, d, M, d, 0, d, 0, 0);
        // agg1 = sum_k f_cat * softmax(scores1)
        att_pool_reduce_kernel<<<(Bb * d * N + 255) / 256, 256, 0, stream>>>(
            CAT1, SCORE, AGG, Bb * d * N);
        // f_agg = att1.mlp(agg1)  (d -> h)
        gemm(att1_mlp[i].w, att1_mlp[i].g, att1_mlp[i].b, AGG, FAGG,
             d, h, N, d, 0, h, 0, 1);
        // f_xyz2 = lfa.mlp2(f_xyz)  (h -> h) -> CAT2 channels [h, d)
        gemm(lfa2[i].w, lfa2[i].g, lfa2[i].b, CAT1, CAT2, h, h, M, d, h, d, h, 1);
        // f_nb2 = gather(f_agg) -> CAT2 channels [0, h)
        gather_nb_kernel<<<(totNK + 255) / 256, 256, 0, stream>>>(
            FAGG, neigh[i], CAT2, h, N, N, Kn, d, 0, totNK);
        // scores2 = att2.fc(f_cat2)
        gemm(att2_fc[i], nullptr, nullptr, CAT2, SCORE, d, d, M, d, 0, d, 0, 0);
        // agg2
        att_pool_reduce_kernel<<<(Bb * d * N + 255) / 256, 256, 0, stream>>>(
            CAT2, SCORE, AGG, Bb * d * N);
        // bb = att2.mlp(agg2)  (d -> d)
        gemm(att2_mlp[i].w, att2_mlp[i].g, att2_mlp[i].b, AGG, BBv,
             d, d, N, d, 0, d, 0, 1);
        // f = mlp2(bb)  (d -> 2d, no act)
        gemm(mlp2_p[i].w, mlp2_p[i].g, mlp2_p[i].b, BBv, T1,
             d, 2 * d, N, d, 0, 2 * d, 0, 0);
        // sc = short(feat)  (din -> 2d, no act)
        gemm(short_p[i].w, short_p[i].g, short_p[i].b, fin, SCb,
             din, 2 * d, N, din, 0, 2 * d, 0, 0);
        // fe = leaky(f + sc)
        float* fe = (i == 0) ? ENCp[0] : FE;
        {
            const size_t n = (size_t)Bb * 2 * d * N;
            add_leaky_kernel<<<(unsigned)((n + 255) / 256), 256, 0, stream>>>(T1, SCb, fe, n);
        }
        // fs = maxpool(fe, sub)
        {
            const int Mn = NSv[i + 1];
            const int tot = Bb * 2 * d * Mn;
            maxpool_kernel<<<(tot + 255) / 256, 256, 0, stream>>>(
                fe, sub[i], ENCp[i + 1], 2 * d, N, Mn, Kn, tot);
        }
    }

    // ---- decoder ----
    // feat = dec0(fs3)  512 -> 512 @ N=128
    gemm(dec0_p.w, dec0_p.g, dec0_p.b, ENCp[4], DF1, 512, 512, 128, 512, 0, 512, 0, 1);
    const float* dprev = DF1;
    int Cfeat = 512, Nin = 128;
    const int encC[5] = { 32, 32, 128, 256, 512 };
    for (int j = 0; j < 4; ++j) {
        const int li   = 3 - j;
        const int Nout = NSv[li];
        const float* skip = ENCp[li];
        const int Cenc = encC[li];
        const int Cin  = Cenc + Cfeat;
        const int Cout = (j < 3) ? 2 * DOUT[2 - j] : 32;
        // concat: skip -> CATIN[0,Cenc) via TDM; interp-gather -> CATIN[Cenc,Cin)
        copy_slab(skip, CATIN, Cenc, Nout, Cin, 0);
        {
            const int tot = Bb * Nout;   // K = 1
            gather_nb_kernel<<<(tot + 255) / 256, 256, 0, stream>>>(
                dprev, interp[li], CATIN, Cfeat, Nout, Nin, 1, Cin, Cenc, tot);
        }
        float* dst = (j < 3) ? ((dprev == DF1) ? DF2 : DF1) : (float*)d_out;
        gemm(dec_p[j].w, dec_p[j].g, dec_p[j].b, CATIN, dst,
             Cin, Cout, Nout, Cin, 0, Cout, 0, 1);
        dprev = dst; Cfeat = Cout; Nin = Nout;
    }
}
